// ScouterAttention_24309514895607
// MI455X (gfx1250) — matrix operand order repaired and checked by reading, not compiled, not run on hardware
//
#include <hip/hip_runtime.h>
#include <hip/hip_bf16.h>
#include <math.h>

// Problem constants (match reference)
#define BB    64
#define NN    4096
#define DD    128
#define SS    20
#define SPAD  32
#define ITERS 3
#define VTR   40   // padded row stride (elements) for transposed v chunk in LDS

typedef __attribute__((ext_vector_type(16))) __bf16 v16bf;
typedef __attribute__((ext_vector_type(8)))  __bf16 v8bf;
typedef __attribute__((ext_vector_type(4)))  __bf16 v4bf;
typedef __attribute__((ext_vector_type(8)))  float  v8f;
typedef __attribute__((ext_vector_type(4)))  float  v4f;

// Native f32 -> bf16 conversion (RNE) — backend emits v_cvt ops instead of
// a software add3/shift rounding sequence.
static __device__ __forceinline__ __bf16 f2bf(float f) { return (__bf16)f; }

static __device__ __forceinline__ v8f wmma_bf16(v16bf a, v16bf b, v8f c) {
  // D = A(16x32 bf16) * B(32x16 bf16) + C(16x16 f32)
  return __builtin_amdgcn_wmma_f32_16x16x32_bf16(
      /*neg_a=*/false, a, /*neg_b=*/false, b,
      /*c_mod=*/(short)0, c, /*reuse_a=*/false, /*reuse_b=*/false);
}

// A fragment: 16x32 bf16 tile, row-major source (row ptr `A`, leading dim lda).
// Lane m=lane&15, half=lane>>4: elems 0..7 = A[m][k0+half*8+i], 8..15 = A[m][k0+16+half*8+i]
static __device__ __forceinline__ v16bf load_a_bf16(const __bf16* A, int lda, int k0, int lane) {
  int m = lane & 15, hh = lane >> 4;
  const __bf16* p = A + (size_t)m * lda + k0 + hh * 8;
  v8bf lo = *(const v8bf*)p;
  v8bf hi = *(const v8bf*)(p + 16);
  return __builtin_shufflevector(lo, hi, 0,1,2,3,4,5,6,7,8,9,10,11,12,13,14,15);
}

// B fragment: 32x16 bf16, where B's column n is contiguous in memory
// (i.e. source is B^T row-major: row n has ldb elems). Lane covers K = k0+half*8+i, k0+16+half*8+i.
static __device__ __forceinline__ v16bf load_bT_bf16(const __bf16* Bt, int ldb, int k0, int lane) {
  int n = lane & 15, hh = lane >> 4;
  const __bf16* p = Bt + (size_t)n * ldb + k0 + hh * 8;
  v8bf lo = *(const v8bf*)p;
  v8bf hi = *(const v8bf*)(p + 16);
  return __builtin_shufflevector(lo, hi, 0,1,2,3,4,5,6,7,8,9,10,11,12,13,14,15);
}

// ---------------- conversion / init kernels ----------------

__global__ void f32_to_bf16_kernel(const float* __restrict__ x, __bf16* __restrict__ y, int n4) {
  int i = blockIdx.x * blockDim.x + threadIdx.x;
  if (i >= n4) return;
  v4f v = ((const v4f*)x)[i];
  v4bf o;
  o[0] = f2bf(v.x); o[1] = f2bf(v.y); o[2] = f2bf(v.z); o[3] = f2bf(v.w);
  ((v4bf*)y)[i] = o;
}

__global__ void init_slots_kernel(const float* __restrict__ init_slots,
                                  float* __restrict__ slots,
                                  __bf16* __restrict__ slotsb,
                                  float* __restrict__ loss) {
  int i = blockIdx.x * blockDim.x + threadIdx.x;        // over BB*SPAD*DD
  if (i >= BB * SPAD * DD) return;
  int d = i % DD;
  int s = (i / DD) % SPAD;
  int b = i / (DD * SPAD);
  float v = (s < SS) ? init_slots[s * DD + d] : 0.0f;
  slotsb[i] = f2bf(v);
  if (s < SS) slots[((size_t)b * SS + s) * DD + d] = v;
  if (i == 0) *loss = 0.0f;
}

__global__ void zero_kernel(float* __restrict__ s2, float* __restrict__ upd,
                            float* __restrict__ loss) {
  int i = blockIdx.x * blockDim.x + threadIdx.x;
  if (i < BB * SPAD) s2[i] = 0.0f;
  if (i < BB * SS * DD) upd[i] = 0.0f;
  if (i == 0) *loss = 0.0f;
}

// ---------------- k = inputs @ Wk^T + bk  (bf16 out) ----------------
// grid: (BB*NN/16) row-tiles, 256 threads = 8 waves; wave w -> 16-column tile w.
__global__ __launch_bounds__(256)
void proj_k_kernel(const float* __restrict__ inp, const __bf16* __restrict__ wkb,
                   const float* __restrict__ bk, __bf16* __restrict__ kout) {
  int mtile = blockIdx.x;
  int wave = threadIdx.x >> 5;
  int lane = threadIdx.x & 31;
  int row0 = mtile * 16;
  int col0 = wave * 16;
  int m = row0 + (lane & 15);
  int hh = lane >> 4;
  v8f acc = {};
  for (int kk = 0; kk < DD; kk += 32) {
    // A frag: convert f32 inputs rows to bf16 on the fly (vector loads + v_cvt)
    const float* pa = inp + (size_t)m * DD + kk + hh * 8;
    v4f a0 = *(const v4f*)(pa);
    v4f a1 = *(const v4f*)(pa + 4);
    v4f a2 = *(const v4f*)(pa + 16);
    v4f a3 = *(const v4f*)(pa + 20);
    v16bf a;
#pragma unroll
    for (int i = 0; i < 4; ++i) a[i]      = f2bf(a0[i]);
#pragma unroll
    for (int i = 0; i < 4; ++i) a[4 + i]  = f2bf(a1[i]);
#pragma unroll
    for (int i = 0; i < 4; ++i) a[8 + i]  = f2bf(a2[i]);
#pragma unroll
    for (int i = 0; i < 4; ++i) a[12 + i] = f2bf(a3[i]);
    // B frag: B[k][n] = Wk[n][k] -> Wk row n contiguous
    v16bf bfr = load_bT_bf16(wkb + (size_t)col0 * DD, DD, kk, lane);
    acc = wmma_bf16(a, bfr, acc);
  }
  int n = col0 + (lane & 15);
  float bias = bk[n];
#pragma unroll
  for (int v = 0; v < 8; ++v) {
    int r = row0 + v + 8 * hh;
    kout[(size_t)r * DD + n] = f2bf(acc[v] + bias);
  }
}

// ---------------- dots[b,s,n] = SCALE * slots[b,s,:] . k[b,n,:] ----------------
// grid: BB * (NN/128), 256 threads; wave w -> 16 n-columns; 2 accs cover 32 padded slots.
__global__ __launch_bounds__(256)
void dots_kernel(const __bf16* __restrict__ kmat, const __bf16* __restrict__ slotsb,
                 float* __restrict__ dots, float* __restrict__ s2) {
  int blk = blockIdx.x;
  int b = blk / (NN / 128);
  int nb = blk % (NN / 128);
  int wave = threadIdx.x >> 5, lane = threadIdx.x & 31;
  int n0 = nb * 128 + wave * 16;
  const __bf16* kb = kmat + ((size_t)b * NN + n0) * DD;     // column base: k rows contiguous
  const __bf16* sb = slotsb + (size_t)b * SPAD * DD;
  v8f acc0 = {}, acc1 = {};
  for (int kk = 0; kk < DD; kk += 32) {
    v16bf a0 = load_a_bf16(sb, DD, kk, lane);               // slots rows 0..15
    v16bf a1 = load_a_bf16(sb + 16 * DD, DD, kk, lane);     // slots rows 16..31 (pad = 0)
    v16bf bf = load_bT_bf16(kb, DD, kk, lane);
    acc0 = wmma_bf16(a0, bf, acc0);
    acc1 = wmma_bf16(a1, bf, acc1);
  }
  const float scale = 0.08838834764831845f;                 // 128^-0.5
  int hh = lane >> 4;
  int n = n0 + (lane & 15);
#pragma unroll
  for (int v = 0; v < 8; ++v) {
    {
      int s = v + 8 * hh;                                   // 0..15, always valid
      float val = acc0[v] * scale;
      dots[((size_t)b * SS + s) * NN + n] = val;
      float r = val;
      for (int o = 1; o < 16; o <<= 1) r += __shfl_xor(r, o, 16);
      if ((lane & 15) == 0) atomicAdd(&s2[b * SPAD + s], r);
    }
    {
      int s = 16 + v + 8 * hh;                              // 16..31
      if (s < SS) {
        float val = acc1[v] * scale;
        dots[((size_t)b * SS + s) * NN + n] = val;
        float r = val;
        for (int o = 1; o < 16; o <<= 1) r += __shfl_xor(r, o, 16);
        if ((lane & 15) == 0) atomicAdd(&s2[b * SPAD + s], r);
      }
    }
  }
}

// ---------------- renorm + softmax*sigmoid -> attn; updates += attn @ v / D ----------------
// grid: BB * (NN/256), 256 threads; K dim = 256 n-values, chunked by 32.
__global__ __launch_bounds__(256)
void attn_updates_kernel(const float* __restrict__ dots, const float* __restrict__ s2,
                         const __bf16* __restrict__ vmat, float* __restrict__ upd,
                         float* __restrict__ lossacc, int last) {
  __shared__ __bf16 attn_lds[SPAD][256];    // [slot][n-local], bf16
  __shared__ __bf16 vch_t[DD][VTR];         // transposed v chunk: [d][n-local], padded stride
  int b = blockIdx.x / (NN / 256);
  int nb = blockIdx.x % (NN / 256);
  int n0 = nb * 256;
  int t = threadIdx.x;

  // ---- phase 1: per-n softmax over the 20 slots ----
  float sv[SS];
  float total = 0.0f;
#pragma unroll
  for (int s = 0; s < SS; ++s) { sv[s] = s2[b * SPAD + s]; total += sv[s]; }
  int n = n0 + t;
  float dd[SS];
  float mx = -1e30f;
#pragma unroll
  for (int s = 0; s < SS; ++s) {
    float x = dots[((size_t)b * SS + s) * NN + n];
    x = x / sv[s] * total;                                  // custom renorm
    dd[s] = x;
    mx = fmaxf(mx, x);
  }
  float se = 0.0f;
#pragma unroll
  for (int s = 0; s < SS; ++s) se += expf(dd[s] - mx);
  float inv_se = 1.0f / se;
  float lsum = 0.0f;
#pragma unroll
  for (int s = 0; s < SS; ++s) {
    float a = expf(dd[s] - mx) * inv_se * (1.0f / (1.0f + expf(-dd[s])));
    attn_lds[s][t] = f2bf(a);
    lsum += a;                                              // relu(a)==a (a>0)
  }
#pragma unroll
  for (int s = SS; s < SPAD; ++s) attn_lds[s][t] = f2bf(0.0f);
  if (last) {
    for (int o = 16; o >= 1; o >>= 1) lsum += __shfl_down(lsum, o, 32);
    if ((t & 31) == 0) atomicAdd(lossacc, lsum);
  }
  __syncthreads();

  // ---- phase 2: updates(32x128) += attn(32x256) @ v(256x128), K chunked by 32 ----
  int wave = t >> 5, lane = t & 31;
  int d0 = wave * 16;
  v8f acc0 = {}, acc1 = {};
  for (int kc = 0; kc < 256; kc += 32) {
    // stage v chunk TRANSPOSED: thread t reads 16 contiguous d of n-row r, scatters to vch_t
    {
      int r = t >> 3, c = (t & 7) * 16;
      const __bf16* src = vmat + ((size_t)b * NN + n0 + kc + r) * DD + c;
      v8bf x0 = *(const v8bf*)(src);
      v8bf x1 = *(const v8bf*)(src + 8);
#pragma unroll
      for (int i = 0; i < 8; ++i) vch_t[c + i][r]     = x0[i];
#pragma unroll
      for (int i = 0; i < 8; ++i) vch_t[c + 8 + i][r] = x1[i];
    }
    __syncthreads();
    // B frag: column d of v chunk is now a contiguous LDS row -> two ds_load_b128 (k = hh*8+i, 16+hh*8+i)
    int dcol = d0 + (lane & 15), hh = lane >> 4;
    v8bf blo = *(const v8bf*)&vch_t[dcol][hh * 8];
    v8bf bhi = *(const v8bf*)&vch_t[dcol][16 + hh * 8];
    v16bf bfr = __builtin_shufflevector(blo, bhi, 0,1,2,3,4,5,6,7,8,9,10,11,12,13,14,15);
    v16bf a0 = load_a_bf16(&attn_lds[0][0], 256, kc, lane);
    v16bf a1 = load_a_bf16(&attn_lds[16][0], 256, kc, lane);
    acc0 = wmma_bf16(a0, bfr, acc0);
    acc1 = wmma_bf16(a1, bfr, acc1);
    __syncthreads();
  }
  const float inv_d = 1.0f / (float)DD;
  int hh = lane >> 4, d = d0 + (lane & 15);
#pragma unroll
  for (int v = 0; v < 8; ++v) {
    int s = v + 8 * hh;
    atomicAdd(&upd[((size_t)b * SS + s) * DD + d], acc0[v] * inv_d);
    int s1 = 16 + v + 8 * hh;
    if (s1 < SS) atomicAdd(&upd[((size_t)b * SS + s1) * DD + d], acc1[v] * inv_d);
  }
}

// ---------------- GRU cell on (BB*SS, DD) rows ----------------
__global__ __launch_bounds__(128)
void gru_kernel(const float* __restrict__ upd, float* __restrict__ slots,
                __bf16* __restrict__ slotsb,
                const float* __restrict__ Wih, const float* __restrict__ Whh,
                const float* __restrict__ bih, const float* __restrict__ bhh) {
  int row = blockIdx.x;            // BB*SS
  int b = row / SS, s = row % SS;
  int t = threadIdx.x;
  __shared__ float xl[DD], hl[DD];
  xl[t] = upd[(size_t)row * DD + t];
  hl[t] = slots[(size_t)row * DD + t];
  __syncthreads();
  float gi[3], gh[3];
#pragma unroll
  for (int g = 0; g < 3; ++g) {
    const float* wi = Wih + (size_t)(g * DD + t) * DD;
    const float* wh = Whh + (size_t)(g * DD + t) * DD;
    float si = 0.0f, sh = 0.0f;
    for (int k = 0; k < DD; ++k) { si += wi[k] * xl[k]; sh += wh[k] * hl[k]; }
    gi[g] = si + bih[g * DD + t];
    gh[g] = sh + bhh[g * DD + t];
  }
  float r = 1.0f / (1.0f + expf(-(gi[0] + gh[0])));
  float z = 1.0f / (1.0f + expf(-(gi[1] + gh[1])));
  float nn2 = tanhf(gi[2] + r * gh[2]);
  float hn = (1.0f - z) * nn2 + z * hl[t];
  slots[(size_t)row * DD + t] = hn;
  slotsb[((size_t)b * SPAD + s) * DD + t] = f2bf(hn);
}

// ---------------- out[b,s] = sum_d updates; out[1280] = loss mean ----------------
__global__ void final_kernel(const float* __restrict__ upd, const float* __restrict__ loss,
                             float* __restrict__ out) {
  int i = blockIdx.x * blockDim.x + threadIdx.x;
  if (i < BB * SS) {
    const float* p = upd + (size_t)i * DD;
    float s = 0.0f;
    for (int d = 0; d < DD; ++d) s += p[d];
    out[i] = s;
  }
  if (i == 0) out[BB * SS] = *loss / (float)((size_t)BB * SS * NN);
}

// ---------------- launch ----------------
extern "C" void kernel_launch(void* const* d_in, const int* in_sizes, int n_in,
                              void* d_out, int out_size, void* d_ws, size_t ws_size,
                              hipStream_t stream) {
  const float* inputs     = (const float*)d_in[0];
  const float* inputs_x   = (const float*)d_in[1];
  const float* init_slots = (const float*)d_in[2];
  const float* Wk         = (const float*)d_in[3];
  const float* bk         = (const float*)d_in[4];
  const float* Wih        = (const float*)d_in[5];
  const float* Whh        = (const float*)d_in[6];
  const float* bih        = (const float*)d_in[7];
  const float* bhh        = (const float*)d_in[8];
  float* out = (float*)d_out;

  char* ws = (char*)d_ws;
  size_t off = 0;
  auto take = [&](size_t bytes) { size_t o = off; off = (off + bytes + 255) & ~(size_t)255; return o; };
  __bf16* vbf    = (__bf16*)(ws + take((size_t)BB * NN * DD * 2));
  __bf16* kbf    = (__bf16*)(ws + take((size_t)BB * NN * DD * 2));
  __bf16* wkb    = (__bf16*)(ws + take((size_t)DD * DD * 2));
  float*  dots   = (float*) (ws + take((size_t)BB * SS * NN * 4));
  float*  s2     = (float*) (ws + take((size_t)BB * SPAD * 4));
  float*  upd    = (float*) (ws + take((size_t)BB * SS * DD * 4));
  float*  slots  = (float*) (ws + take((size_t)BB * SS * DD * 4));
  __bf16* slotsb = (__bf16*)(ws + take((size_t)BB * SPAD * DD * 2));
  float*  loss   = (float*) (ws + take(256));

  // bf16 conversions (v, Wk) + slot init
  f32_to_bf16_kernel<<<(BB * NN * DD / 4 + 255) / 256, 256, 0, stream>>>(inputs_x, vbf, BB * NN * DD / 4);
  f32_to_bf16_kernel<<<(DD * DD / 4 + 255) / 256, 256, 0, stream>>>(Wk, wkb, DD * DD / 4);
  init_slots_kernel<<<(BB * SPAD * DD + 255) / 256, 256, 0, stream>>>(init_slots, slots, slotsb, loss);

  // k projection (WMMA)
  proj_k_kernel<<<BB * NN / 16, 256, 0, stream>>>(inputs, wkb, bk, kbf);

  for (int it = 0; it < ITERS; ++it) {
    zero_kernel<<<(BB * SS * DD + 255) / 256, 256, 0, stream>>>(s2, upd, loss);
    dots_kernel<<<BB * (NN / 128), 256, 0, stream>>>(kbf, slotsb, dots, s2);
    attn_updates_kernel<<<BB * (NN / 256), 256, 0, stream>>>(dots, s2, vbf, upd, loss,
                                                             it == ITERS - 1 ? 1 : 0);
    gru_kernel<<<BB * SS, 128, 0, stream>>>(upd, slots, slotsb, Wih, Whh, bih, bhh);
  }

  final_kernel<<<(BB * SS + 255) / 256, 256, 0, stream>>>(upd, loss, out);
}
